// Net_39556648796609
// MI455X (gfx1250) — compile-verified, matrix operand order checked by reading, not run on hardware
//
#include <hip/hip_runtime.h>
#include <hip/hip_bf16.h>

#define NN   50000
#define EE   800000
#define KK   25
#define CIN  3
#define COUT 32
#define EPSF 1e-15f
#define NTILES 3125   // NN / 16 exactly
#define HST  72       // f16 LDS row stride: 144B rows -> b128-aligned, conflict-free

typedef __attribute__((ext_vector_type(16))) _Float16 v16h;
typedef __attribute__((ext_vector_type(8)))  _Float16 v8h;
typedef __attribute__((ext_vector_type(8)))  float    v8f;

// ---------------------------------------------------------------- init ws ---
__global__ __launch_bounds__(256) void init_ws(float* __restrict__ p, int n) {
  int i = blockIdx.x * 256 + threadIdx.x;
  if (i < n) p[i] = 0.f;
}

// ------------------------------------------------------ edge (GMM message) ---
// One thread per edge. g / mu / 1/(eps+sigma^2) staged in LDS (uniform reads ->
// DS broadcast). Scatter via global_atomic_add_f32 into agg (L2 resident).
__global__ __launch_bounds__(256) void gmm_edge(
    const float* __restrict__ x, const float* __restrict__ pos,
    const int* __restrict__ ei, const float* __restrict__ g,
    const float* __restrict__ mu, const float* __restrict__ sigma,
    float* __restrict__ agg, float* __restrict__ deg)
{
  __shared__ float sg[CIN * KK * COUT];   // g[c][k][m]
  __shared__ float smu[KK * 3];
  __shared__ float sinv[KK * 3];          // 1/(eps + sigma^2)
  int tid = threadIdx.x;
  for (int i = tid; i < CIN * KK * COUT; i += 256) sg[i] = g[i];
  for (int i = tid; i < KK * 3; i += 256) {
    smu[i] = mu[i];
    float s = sigma[i];
    sinv[i] = 1.f / (EPSF + s * s);
  }
  __syncthreads();

  int e = blockIdx.x * 256 + tid;
  if (e >= EE) return;
  int src = ei[e], dst = ei[EE + e];

  float p0 = pos[dst*3+0] - pos[src*3+0];
  float p1 = pos[dst*3+1] - pos[src*3+1];
  float p2 = pos[dst*3+2] - pos[src*3+2];
  float x0 = x[src*3+0], x1 = x[src*3+1], x2 = x[src*3+2];

  float acc[COUT];
  #pragma unroll
  for (int m = 0; m < COUT; ++m) acc[m] = 0.f;

  #pragma unroll 1
  for (int k = 0; k < KK; ++k) {
    float d0 = p0 - smu[k*3+0];
    float d1 = p1 - smu[k*3+1];
    float d2 = p2 - smu[k*3+2];
    float q  = d0*d0*sinv[k*3+0] + d1*d1*sinv[k*3+1] + d2*d2*sinv[k*3+2];
    float gk = __expf(-0.5f * q);
    float w0 = gk * x0, w1 = gk * x1, w2 = gk * x2;
    const float* g0 = &sg[k * COUT];
    const float* g1 = &sg[KK * COUT + k * COUT];
    const float* g2 = &sg[2 * KK * COUT + k * COUT];
    #pragma unroll
    for (int m = 0; m < COUT; ++m)
      acc[m] += w0 * g0[m] + w1 * g1[m] + w2 * g2[m];
  }

  float* ag = &agg[(size_t)dst * COUT];
  #pragma unroll
  for (int m = 0; m < COUT; ++m) atomicAdd(&ag[m], acc[m]);
  atomicAdd(&deg[dst], 1.f);
}

// ------------------------------------------------------ WMMA fragment utils ---
// Both A and B fragments read two contiguous 8xf16 runs (K = kb..kb+7 and
// 16+kb..16+kb+7, kb = 8*(lane>=16)) from an f16 row of stride HST.
// 144B rows keep every run 16B-aligned -> two ds_load_b128 per fragment.
__device__ __forceinline__ v16h frag_ld(const _Float16* p) {
  v8h lo = *(const v8h*)p;
  v8h hi = *(const v8h*)(p + 16);
  return __builtin_shufflevector(lo, hi, 0,1,2,3,4,5,6,7,8,9,10,11,12,13,14,15);
}
// A (16x32): lane -> row lane%16, K chunk at kc
__device__ __forceinline__ v16h frag_a(const _Float16* h, int kc, int lane) {
  int kb = (lane & 16) ? 8 : 0;
  return frag_ld(h + (lane & 15) * HST + kc + kb);
}
// B (32x16) from LDS-transposed weights Wt[n][k] (stride HST), cols cb..cb+15
__device__ __forceinline__ v16h frag_b(const _Float16* Wt, int kc, int cb, int lane) {
  int kb = (lane & 16) ? 8 : 0;
  return frag_ld(Wt + (cb + (lane & 15)) * HST + kc + kb);
}
__device__ __forceinline__ v8f wmma_f16(v16h a, v16h b, v8f c) {
  return __builtin_amdgcn_wmma_f32_16x16x32_f16(false, a, false, b, (short)0, c, false, false);
}

// ------------------------------------------------- node update + MLP head ---
// 8 waves / block, each wave owns a 16-node tile. 9 v_wmma per wave.
__global__ __launch_bounds__(256) void mlp_head(
    const float* __restrict__ x, const float* __restrict__ agg,
    const float* __restrict__ deg,
    const float* __restrict__ rw,  const float* __restrict__ cbias,
    const float* __restrict__ Wpt, const float* __restrict__ bpt,
    const float* __restrict__ Wl1, const float* __restrict__ bl1,
    const float* __restrict__ Wl2, const float* __restrict__ bl2,
    const float* __restrict__ Wl3, const float* __restrict__ bl3,
    const float* __restrict__ Wl4, const float* __restrict__ bl4,
    float* __restrict__ out)
{
  // f16 transposed weights: Wt[n*HST + k]
  __shared__ __align__(16) _Float16 sWptT[64 * HST];   // [n=64][k=32]
  __shared__ __align__(16) _Float16 sWl1T[32 * HST];   // [n=32][k=64]
  __shared__ __align__(16) _Float16 sWl2T[16 * HST];   // [n=16][k=32]
  __shared__ float sWl3[16*4], sWl4[16*3];
  __shared__ float sbpt[64], sbl1[32], sbl2[16], sbl3[4], sbl4[3];
  __shared__ float srw[96], scb[32];
  __shared__ __align__(16) _Float16 hb_all[8][16 * HST];  // per-wave f16 tile

  int tid = threadIdx.x;
  for (int i = tid; i < 32*64; i += 256) { int k = i >> 6, n = i & 63; sWptT[n*HST + k] = (_Float16)Wpt[i]; }
  for (int i = tid; i < 64*32; i += 256) { int k = i >> 5, n = i & 31; sWl1T[n*HST + k] = (_Float16)Wl1[i]; }
  for (int i = tid; i < 32*16; i += 256) { int k = i >> 4, n = i & 15; sWl2T[n*HST + k] = (_Float16)Wl2[i]; }
  for (int i = tid; i < 96;    i += 256) srw[i] = rw[i];
  if (tid < 64) { sWl3[tid] = Wl3[tid]; sbpt[tid] = bpt[tid]; }
  if (tid < 48)   sWl4[tid] = Wl4[tid];
  if (tid < 32) { scb[tid] = cbias[tid]; sbl1[tid] = bl1[tid]; }
  if (tid < 16)   sbl2[tid] = bl2[tid];
  if (tid < 4)    sbl3[tid] = bl3[tid];
  if (tid < 3)    sbl4[tid] = bl4[tid];
  __syncthreads();

  int wv = tid >> 5, lane = tid & 31;
  int tile = blockIdx.x * 8 + wv;
  if (tile >= NTILES) tile = NTILES - 1;  // tail waves redo last tile (identical writes)
  int nb = tile * 16;
  _Float16* hb = hb_all[wv];
  int col = lane & 15;
  int rb  = (lane & 16) ? 8 : 0;

  // ---- h0 = relu(agg/deg + x @ root_w + conv_b)  [16 x 32], lane = channel
  for (int j = 0; j < 16; ++j) {
    int n = nb + j;
    float v = agg[(size_t)n * 32 + lane] / fmaxf(deg[n], 1.f);
    v += srw[lane] * x[n*3+0] + srw[32+lane] * x[n*3+1] + srw[64+lane] * x[n*3+2];
    v += scb[lane];
    hb[j * HST + lane] = (_Float16)fmaxf(v, 0.f);
  }
  __syncthreads();

  // ---- layer pt: [16,32] @ [32,64]  (4 WMMAs)
  v16h a0 = frag_a(hb, 0, lane);
  v8f cc[4];
  #pragma unroll
  for (int t = 0; t < 4; ++t) {
    v8f c; float bv = sbpt[t*16 + col];
    #pragma unroll
    for (int r = 0; r < 8; ++r) c[r] = bv;
    cc[t] = wmma_f16(a0, frag_b(sWptT, 0, t*16, lane), c);
  }
  __syncthreads();
  #pragma unroll
  for (int t = 0; t < 4; ++t)
    #pragma unroll
    for (int r = 0; r < 8; ++r)
      hb[(r + rb) * HST + t*16 + col] = (_Float16)fmaxf(cc[t][r], 0.f);
  __syncthreads();

  // ---- layer l1: [16,64] @ [64,32]  (2 K-chunks x 2 N-tiles = 4 WMMAs)
  v16h a1 = frag_a(hb, 0, lane);
  v16h a2 = frag_a(hb, 32, lane);
  v8f dd[2];
  #pragma unroll
  for (int t = 0; t < 2; ++t) {
    v8f c; float bv = sbl1[t*16 + col];
    #pragma unroll
    for (int r = 0; r < 8; ++r) c[r] = bv;
    c = wmma_f16(a1, frag_b(sWl1T, 0,  t*16, lane), c);
    c = wmma_f16(a2, frag_b(sWl1T, 32, t*16, lane), c);
    dd[t] = c;
  }
  __syncthreads();
  #pragma unroll
  for (int t = 0; t < 2; ++t)
    #pragma unroll
    for (int r = 0; r < 8; ++r)
      hb[(r + rb) * HST + t*16 + col] = (_Float16)fmaxf(dd[t][r], 0.f);
  __syncthreads();

  // ---- layer l2: [16,32] @ [32,16]  (1 WMMA)
  v16h a3 = frag_a(hb, 0, lane);
  v8f c2; { float bv = sbl2[col];
    #pragma unroll
    for (int r = 0; r < 8; ++r) c2[r] = bv; }
  c2 = wmma_f16(a3, frag_b(sWl2T, 0, 0, lane), c2);
  __syncthreads();
  #pragma unroll
  for (int r = 0; r < 8; ++r)
    hb[(r + rb) * HST + col] = (_Float16)fmaxf(c2[r], 0.f);
  __syncthreads();

  // ---- heads: 16->4 and 16->3 + log_softmax, lane-per-node VALU
  if (lane < 16) {
    const _Float16* row = &hb[lane * HST];
    int n = nb + lane;
    float z3[4];
    #pragma unroll
    for (int j = 0; j < 4; ++j) {
      float s = sbl3[j];
      #pragma unroll
      for (int i = 0; i < 16; ++i) s += (float)row[i] * sWl3[i*4 + j];
      z3[j] = s;
    }
    float mx = fmaxf(fmaxf(z3[0], z3[1]), fmaxf(z3[2], z3[3]));
    float se = __expf(z3[0]-mx) + __expf(z3[1]-mx) + __expf(z3[2]-mx) + __expf(z3[3]-mx);
    float ls = mx + __logf(se);
    #pragma unroll
    for (int j = 0; j < 4; ++j) out[(size_t)n * 4 + j] = z3[j] - ls;

    float z4[3];
    #pragma unroll
    for (int j = 0; j < 3; ++j) {
      float s = sbl4[j];
      #pragma unroll
      for (int i = 0; i < 16; ++i) s += (float)row[i] * sWl4[i*3 + j];
      z4[j] = s;
    }
    float mx2 = fmaxf(fmaxf(z4[0], z4[1]), z4[2]);
    float se2 = __expf(z4[0]-mx2) + __expf(z4[1]-mx2) + __expf(z4[2]-mx2);
    float ls2 = mx2 + __logf(se2);
    #pragma unroll
    for (int j = 0; j < 3; ++j) out[(size_t)NN * 4 + (size_t)n * 3 + j] = z4[j] - ls2;
  }
}

// ----------------------------------------------------------------- launch ---
extern "C" void kernel_launch(void* const* d_in, const int* in_sizes, int n_in,
                              void* d_out, int out_size, void* d_ws, size_t ws_size,
                              hipStream_t stream) {
  const float* x      = (const float*)d_in[0];
  const float* pos    = (const float*)d_in[1];
  const int*   ei     = (const int*)  d_in[2];
  // d_in[3] = batch (unused)
  const float* g      = (const float*)d_in[4];
  const float* mu     = (const float*)d_in[5];
  const float* sigma  = (const float*)d_in[6];
  const float* root_w = (const float*)d_in[7];
  const float* conv_b = (const float*)d_in[8];
  const float* Wpt    = (const float*)d_in[9];
  const float* bpt    = (const float*)d_in[10];
  const float* Wl1    = (const float*)d_in[11];
  const float* bl1    = (const float*)d_in[12];
  const float* Wl2    = (const float*)d_in[13];
  const float* bl2    = (const float*)d_in[14];
  const float* Wl3    = (const float*)d_in[15];
  const float* bl3    = (const float*)d_in[16];
  const float* Wl4    = (const float*)d_in[17];
  const float* bl4    = (const float*)d_in[18];

  float* agg = (float*)d_ws;               // [N, 32]
  float* deg = agg + (size_t)NN * 32;      // [N]

  int zn = NN * 33;
  init_ws<<<(zn + 255) / 256, 256, 0, stream>>>(agg, zn);

  gmm_edge<<<(EE + 255) / 256, 256, 0, stream>>>(x, pos, ei, g, mu, sigma, agg, deg);

  int blocks = (NTILES + 7) / 8;           // 8 wave-tiles per 256-thread block
  mlp_head<<<blocks, 256, 0, stream>>>(x, agg, deg, root_w, conv_b,
                                       Wpt, bpt, Wl1, bl1, Wl2, bl2,
                                       Wl3, bl3, Wl4, bl4, (float*)d_out);
}
